// MultiheadfAttention_43361989820624
// MI455X (gfx1250) — matrix operand order repaired and checked by reading, not compiled, not run on hardware
//
#include <hip/hip_runtime.h>
#include <hip/hip_bf16.h>
#include <math.h>

// Problem constants (from reference): B=2, T=2048, C=512, H=8, HD=64, TOPK=64
#define BB 2
#define TT 2048
#define CC 512
#define HH 8
#define HD 64
#define TOPK 64

#define KEY_NEG_INF 0x007FFFFFu  // f2key(-inf)

typedef __attribute__((ext_vector_type(16))) __bf16 v16bf;
typedef __attribute__((ext_vector_type(8)))  float  v8f;

// ---------------------------------------------------------------------------
// WMMA operand loaders for v_wmma_f32_16x16x32_bf16 (wave32)
// A-matrix 16x32 (MxK), contiguous k order used by this code (matches load_bt16):
//   lane L (0..15): M=L, holds K=0..15; lane L+16: M=L, K=16..31
// ---------------------------------------------------------------------------
__device__ __forceinline__ v16bf load_a16(const __bf16* __restrict__ src, int ld) {
  const int lane = threadIdx.x & 31;
  const int m    = lane & 15;
  const int sel  = lane >> 4;
  const __bf16* p = src + (size_t)m * ld;
  v16bf a;
#pragma unroll
  for (int v = 0; v < 8; ++v) {
    const int k = (sel << 4) + (v << 1);
    a[2 * v]     = p[k];
    a[2 * v + 1] = p[k + 1];
  }
  return a;
}

// B-matrix 32x16 (KxN) supplied as its TRANSPOSE: src is N x K row-major
// (src[n*ld + k]).  lanes 0..15: N=lane, K=0..15; lanes 16..31: N=lane-16,
// K=16..31.  Fully contiguous 16 bf16 per lane -> 2x global_load_b128.
__device__ __forceinline__ v16bf load_bt16(const __bf16* __restrict__ src, int ld) {
  const int lane  = threadIdx.x & 31;
  const int n     = lane & 15;
  const int khalf = lane >> 4;
  const __bf16* p = src + (size_t)n * ld + (khalf << 4);
  v16bf b;
#pragma unroll
  for (int v = 0; v < 8; ++v) {
    b[2 * v]     = p[2 * v];
    b[2 * v + 1] = p[2 * v + 1];
  }
  return b;
}

__device__ __forceinline__ v8f wmma_bf16(v16bf a, v16bf b, v8f c) {
  return __builtin_amdgcn_wmma_f32_16x16x32_bf16(false, a, false, b, (short)0, c,
                                                 false, false);
}

// Order-preserving float -> uint32 key (monotonic, handles +-inf)
__device__ __forceinline__ unsigned f2key(float f) {
  unsigned u = __float_as_uint(f);
  return (u & 0x80000000u) ? ~u : (u | 0x80000000u);
}
__device__ __forceinline__ float key2f(unsigned k) {
  unsigned u = (k & 0x80000000u) ? (k & 0x7FFFFFFFu) : ~k;
  return __uint_as_float(u);
}

// ---------------------------------------------------------------------------
// Per-row exact top-64 threshold + unnormalized softmax, templated on the
// number of score elements per lane (NV*32 columns scanned, NV*32 >= Vr).
// Columns >= Vcols are synthesized as -inf; exp written in place as bf16.
// ---------------------------------------------------------------------------
template <int NV>
__device__ __forceinline__ void softmax_rows(float* S, float* sums, int Vcols,
                                             int wave, int lane) {
  for (int rr = 0; rr < 2; ++rr) {
    const int r = wave * 2 + rr;
    unsigned key[NV];
#pragma unroll
    for (int i = 0; i < NV; ++i) {
      const int col = i * 32 + lane;
      key[i] = (col < Vcols) ? f2key(S[r * TT + col]) : KEY_NEG_INF;
    }

    unsigned kmax = 0u;
#pragma unroll
    for (int i = 0; i < NV; ++i) kmax = (key[i] > kmax) ? key[i] : kmax;
#pragma unroll
    for (int off = 16; off; off >>= 1) {
      unsigned o = (unsigned)__shfl_xor((int)kmax, off, 32);
      kmax = (o > kmax) ? o : kmax;
    }
    const float fmaxv = key2f(kmax);

    // binary search: largest key t with count(key >= t) >= TOPK (== 64th largest)
    unsigned lo = 0u, hi = 0xFFFFFFFFu;
    while (lo < hi) {
      const unsigned mid = lo + ((hi - lo) >> 1) + 1u;  // in (lo, hi]
      int c = 0;
#pragma unroll
      for (int i = 0; i < NV; ++i) c += (key[i] >= mid) ? 1 : 0;
#pragma unroll
      for (int off = 16; off; off >>= 1) c += __shfl_xor(c, off, 32);
      if (c >= TOPK) lo = mid; else hi = mid - 1u;
    }

    // unnormalized exp written in place as bf16 (1/sum folded into output)
    float sum = 0.f;
    __bf16* Prow = (__bf16*)(S + r * TT);
#pragma unroll
    for (int i = 0; i < NV; ++i) {
      float e = 0.f;
      if (key[i] >= lo) e = __expf(key2f(key[i]) - fmaxv);
      sum += e;
      Prow[i * 32 + lane] = (__bf16)e;
    }
#pragma unroll
    for (int off = 16; off; off >>= 1) sum += __shfl_xor(sum, off, 32);
    if (lane == 0) sums[r] = sum;
  }
}

// ---------------------------------------------------------------------------
// fp32 -> bf16 conversion (activations, row-major preserved)
// ---------------------------------------------------------------------------
__global__ void cvt_f32_bf16(const float* __restrict__ src, __bf16* __restrict__ dst,
                             int n) {
  int i = blockIdx.x * blockDim.x + threadIdx.x;
  if (i < n) dst[i] = (__bf16)src[i];
}

// ---------------------------------------------------------------------------
// fp32 [K][N] -> bf16 [N][K] transposed convert (weights), LDS-tiled 32x32.
// block = (32,8), grid = (K/32, N/32).
// ---------------------------------------------------------------------------
__global__ void cvt_transpose_kernel(const float* __restrict__ src,
                                     __bf16* __restrict__ dst) {
  __shared__ __bf16 tile[32][33];
  const int k0 = blockIdx.x * 32;
  const int n0 = blockIdx.y * 32;
  const int tx = threadIdx.x;   // 0..31
  const int ty = threadIdx.y;   // 0..7
#pragma unroll
  for (int j = 0; j < 4; ++j) {
    const int k = ty + j * 8;
    tile[k][tx] = (__bf16)src[(size_t)(k0 + k) * CC + n0 + tx];
  }
  __syncthreads();
#pragma unroll
  for (int j = 0; j < 4; ++j) {
    const int n = ty + j * 8;
    dst[(size_t)(n0 + n) * CC + k0 + tx] = tile[tx][n];
  }
}

// ---------------------------------------------------------------------------
// Fused QKV projection. grid = (256 row tiles, 8 heads, 3 mats), block = 32.
// One wave computes a 16x64 tile of (X @ W + b).  Weights arrive TRANSPOSED
// (WT[n][k]) so B operands load contiguously.
//   mat 0 (Q): scale by 1/sqrt(HD), store Qh [b,h,t,d] bf16
//   mat 1 (K): store Kh [b,h,t,d] bf16
//   mat 2 (V): per-row L2 normalize over HD, store TRANSPOSED VhT [b,h,d,t]
// ---------------------------------------------------------------------------
__global__ void qkv_proj_kernel(const __bf16* __restrict__ X,
                                const __bf16* __restrict__ WqT,
                                const __bf16* __restrict__ WkT,
                                const __bf16* __restrict__ WvT,
                                const float* __restrict__ bq,
                                const float* __restrict__ bk,
                                const float* __restrict__ bv,
                                __bf16* __restrict__ Qh, __bf16* __restrict__ Kh,
                                __bf16* __restrict__ VhT) {
  const int tile = blockIdx.x;   // 0..255
  const int head = blockIdx.y;   // 0..7
  const int mat  = blockIdx.z;   // 0=Q 1=K 2=V
  const __bf16* WT   = (mat == 0) ? WqT : (mat == 1) ? WkT : WvT;
  const float*  bias = (mat == 0) ? bq : (mat == 1) ? bk : bv;
  __bf16*       Out  = (mat == 0) ? Qh : (mat == 1) ? Kh : VhT;

  const int row0 = tile * 16;
  __shared__ float tileLds[16 * 64];
  __shared__ float rsc[16];

  v8f acc[4] = {};
  for (int kk = 0; kk < CC; kk += 32) {
    v16bf a = load_a16(X + (size_t)row0 * CC + kk, CC);
#pragma unroll
    for (int nt = 0; nt < 4; ++nt) {
      v16bf b = load_bt16(WT + (size_t)(head * HD + nt * 16) * CC + kk, CC);
      acc[nt] = wmma_bf16(a, b, acc[nt]);
    }
  }

  const int lane = threadIdx.x & 31;
  const int n    = lane & 15;
  const int sel  = lane >> 4;
#pragma unroll
  for (int nt = 0; nt < 4; ++nt)
#pragma unroll
    for (int j = 0; j < 8; ++j) {
      const int m   = j + (sel << 3);
      const int col = head * HD + nt * 16 + n;
      tileLds[m * 64 + nt * 16 + n] = acc[nt][j] + bias[col];
    }
  __syncthreads();

  if (mat == 2 && lane < 16) {
    float s = 0.f;
#pragma unroll
    for (int d = 0; d < 64; ++d) {
      float v = tileLds[lane * 64 + d];
      s += v * v;
    }
    rsc[lane] = 1.0f / fmaxf(sqrtf(s), 1e-12f);
  }
  __syncthreads();

  const float qscale = (mat == 0) ? 0.125f : 1.0f;  // 1/sqrt(64)
  for (int idx = lane; idx < 16 * 64; idx += 32) {
    const int m = idx >> 6;
    const int d = idx & 63;
    float v = tileLds[idx] * qscale;
    const int r  = row0 + m;      // flat (b*T + t)
    const int bi = r >> 11;       // / 2048
    const int ti = r & 2047;
    if (mat == 2) {
      v *= rsc[m];
      Out[(((size_t)bi * HH + head) * HD + d) * TT + ti] = (__bf16)v;  // [b,h,d,t]
    } else {
      Out[(((size_t)bi * HH + head) * TT + ti) * HD + d] = (__bf16)v;  // [b,h,t,d]
    }
  }
}

// ---------------------------------------------------------------------------
// Attention: grid = (T/16, B*H), block = 256 (8 waves).
// LDS: S[16][2048] fp32 scores (exp written back in place as bf16),
//      O1/O2[16][64] fp32 partials, sums[16].
// All phases operate only on the causally-valid column range.
// ---------------------------------------------------------------------------
#define ATTN_SMEM (16 * TT * 4 + 2 * 16 * HD * 4 + 16 * 4)

__global__ void attn_kernel(const __bf16* __restrict__ Qh,
                            const __bf16* __restrict__ Kh,
                            const __bf16* __restrict__ VhT,
                            __bf16* __restrict__ Y) {
  extern __shared__ char smem[];
  float* S    = (float*)smem;                      // 16*2048 fp32
  float* O1   = (float*)(smem + 16 * TT * 4);      // 16*64
  float* O2   = O1 + 16 * HD;                      // 16*64
  float* sums = O2 + 16 * HD;                      // 16

  const int qt   = blockIdx.x;       // query tile 0..127
  const int bh   = blockIdx.y;       // b*H + h, 0..15
  const int q0   = qt * 16;
  const int tid  = threadIdx.x;
  const int wave = tid >> 5;
  const int lane = tid & 31;

  const int Vcols = q0 + 16;                 // causally valid columns
  const int Vr    = (Vcols + 31) & ~31;      // rounded to WMMA K-step

  const __bf16* Qp = Qh  + (size_t)bh * TT * HD;
  const __bf16* Kp = Kh  + (size_t)bh * TT * HD;
  const __bf16* Vp = VhT + (size_t)bh * HD * TT;   // [d][t]

  // zero the output partials (some phase-3 wave ranges can be empty)
  for (int i = tid; i < 2 * 16 * HD; i += 256) O1[i] = 0.f;

  // ---- Phase 1: S = (Q * scale) @ K^T with causal mask (skip tiles past diag)
  {
    v16bf a0 = load_a16(Qp + (size_t)q0 * HD, HD);       // d = 0..31
    v16bf a1 = load_a16(Qp + (size_t)q0 * HD + 32, HD);  // d = 32..63
    for (int kt = wave; kt <= qt; kt += 8) {
      const int key0 = kt * 16;
      // prefetch next K tile for this wave (global_prefetch_b8)
      const int ktn = kt + 8;
      if (ktn <= qt)
        __builtin_prefetch(Kp + (size_t)(ktn * 16 + (lane & 15)) * HD, 0, 3);
      v16bf b0 = load_bt16(Kp + (size_t)key0 * HD, HD);       // K^T, d = 0..31
      v16bf b1 = load_bt16(Kp + (size_t)key0 * HD + 32, HD);  // d = 32..63
      v8f acc = {};
      acc = wmma_bf16(a0, b0, acc);
      acc = wmma_bf16(a1, b1, acc);
      const int n   = lane & 15;
      const int sel = lane >> 4;
#pragma unroll
      for (int j = 0; j < 8; ++j) {
        const int m  = j + (sel << 3);
        const int gk = key0 + n;
        S[m * TT + gk] = (gk <= q0 + m) ? acc[j] : -__builtin_inff();
      }
    }
  }
  __syncthreads();

  // ---- Phase 2: top-64 + softmax over valid columns only (size-class dispatch)
  {
    const int nv = Vr >> 5;                       // elements/lane actually needed
    const int cls = ((nv + 15) >> 4) << 4;        // 16 / 32 / 48 / 64
    if (cls == 16)      softmax_rows<16>(S, sums, Vcols, wave, lane);
    else if (cls == 32) softmax_rows<32>(S, sums, Vcols, wave, lane);
    else if (cls == 48) softmax_rows<48>(S, sums, Vcols, wave, lane);
    else                softmax_rows<64>(S, sums, Vcols, wave, lane);
  }
  __syncthreads();

  // ---- Phase 3: O = P @ V over valid K range only.
  // wave -> (N-tile, K-half); B operand from VhT is contiguous per lane.
  {
    const int nt    = wave & 3;   // N tile (16 of 64 dims)
    const int khalf = wave >> 2;  // half of valid K range
    const int hB    = ((Vr >> 1) + 31) & ~31;
    const int kbeg  = khalf ? hB : 0;
    const int kend  = khalf ? Vr : hB;
    v8f acc = {};
    const __bf16* Pbase = (const __bf16*)S;  // row stride: 2048 fp32 = 4096 bf16
    for (int k0 = kbeg; k0 < kend; k0 += 32) {
      v16bf a = load_a16(Pbase + k0, 2 * TT);
      v16bf b = load_bt16(Vp + (size_t)(nt * 16) * TT + k0, TT);
      acc = wmma_bf16(a, b, acc);
    }
    if (kbeg < kend) {
      float* Obuf = khalf ? O2 : O1;
      const int n   = lane & 15;
      const int sel = lane >> 4;
#pragma unroll
      for (int j = 0; j < 8; ++j) {
        const int m = j + (sel << 3);
        Obuf[m * HD + nt * 16 + n] = acc[j];
      }
    }
  }
  __syncthreads();

  {
    const int bi = bh >> 3;
    const int h  = bh & 7;
    for (int idx = tid; idx < 16 * HD; idx += 256) {
      const int m = idx >> 6;
      const int d = idx & 63;
      const float v = (O1[idx] + O2[idx]) / sums[m];
      Y[((size_t)bi * TT + q0 + m) * CC + h * HD + d] = (__bf16)v;
    }
  }
}

// ---------------------------------------------------------------------------
// Output projection: Out(fp32) = Y(bf16) @ Wp(bf16) + bp, Wp transposed.
// grid = (256 row tiles, 8 col groups of 64), block = 32.
// ---------------------------------------------------------------------------
__global__ void out_proj_kernel(const __bf16* __restrict__ Yb,
                                const __bf16* __restrict__ WpT,
                                const float* __restrict__ bp,
                                float* __restrict__ Out) {
  const int tile = blockIdx.x;
  const int cg   = blockIdx.y;
  const int row0 = tile * 16;

  v8f acc[4] = {};
  for (int kk = 0; kk < CC; kk += 32) {
    v16bf a = load_a16(Yb + (size_t)row0 * CC + kk, CC);
#pragma unroll
    for (int nt = 0; nt < 4; ++nt) {
      v16bf b = load_bt16(WpT + (size_t)(cg * 64 + nt * 16) * CC + kk, CC);
      acc[nt] = wmma_bf16(a, b, acc[nt]);
    }
  }

  const int lane = threadIdx.x & 31;
  const int n    = lane & 15;
  const int sel  = lane >> 4;
#pragma unroll
  for (int nt = 0; nt < 4; ++nt)
#pragma unroll
    for (int j = 0; j < 8; ++j) {
      const int m   = j + (sel << 3);
      const int col = cg * 64 + nt * 16 + n;
      Out[(size_t)(row0 + m) * CC + col] = acc[nt][j] + bp[col];
    }
}

// ---------------------------------------------------------------------------
// Host launcher
// ---------------------------------------------------------------------------
extern "C" void kernel_launch(void* const* d_in, const int* in_sizes, int n_in,
                              void* d_out, int out_size, void* d_ws, size_t ws_size,
                              hipStream_t stream) {
  (void)in_sizes; (void)n_in; (void)out_size; (void)ws_size;

  const float* q  = (const float*)d_in[0];
  // d_in[1] = tgt_mask (causal mask implemented analytically)
  const float* Wq = (const float*)d_in[2];
  const float* bq = (const float*)d_in[3];
  const float* Wk = (const float*)d_in[4];
  const float* bk = (const float*)d_in[5];
  const float* Wv = (const float*)d_in[6];
  const float* bv = (const float*)d_in[7];
  const float* Wp = (const float*)d_in[8];
  const float* bp = (const float*)d_in[9];
  float* out = (float*)d_out;

  // Workspace layout (bf16 buffers), total ~22 MB
  char* ws = (char*)d_ws;
  __bf16* qb   = (__bf16*)(ws);                          // 4 MB: q bf16 [B*T, C]
  __bf16* WqT  = (__bf16*)(ws + (4u  << 20));            // 512 KB each, transposed
  __bf16* WkT  = (__bf16*)(ws + (4u  << 20) + (512u << 10));
  __bf16* WvT  = (__bf16*)(ws + (5u  << 20));
  __bf16* WpT  = (__bf16*)(ws + (5u  << 20) + (512u << 10));
  __bf16* Qh   = (__bf16*)(ws + (6u  << 20));            // 4 MB [B,H,T,HD]
  __bf16* Kh   = (__bf16*)(ws + (10u << 20));            // 4 MB [B,H,T,HD]
  __bf16* VhT  = (__bf16*)(ws + (14u << 20));            // 4 MB [B,H,HD,T]
  __bf16* Yb   = (__bf16*)(ws + (18u << 20));            // 4 MB [B*T, C]

  const int nQ = BB * TT * CC;   // 2,097,152

  cvt_f32_bf16<<<nQ / 256, 256, 0, stream>>>(q, qb, nQ);

  dim3 tb(32, 8), tg(CC / 32, CC / 32);
  cvt_transpose_kernel<<<tg, tb, 0, stream>>>(Wq, WqT);
  cvt_transpose_kernel<<<tg, tb, 0, stream>>>(Wk, WkT);
  cvt_transpose_kernel<<<tg, tb, 0, stream>>>(Wv, WvT);
  cvt_transpose_kernel<<<tg, tb, 0, stream>>>(Wp, WpT);

  qkv_proj_kernel<<<dim3(BB * TT / 16, HH, 3), 32, 0, stream>>>(
      qb, WqT, WkT, WvT, bq, bk, bv, Qh, Kh, VhT);

  (void)hipFuncSetAttribute((const void*)attn_kernel,
                            hipFuncAttributeMaxDynamicSharedMemorySize, ATTN_SMEM);
  attn_kernel<<<dim3(TT / 16, BB * HH), 256, ATTN_SMEM, stream>>>(Qh, Kh, VhT, Yb);

  out_proj_kernel<<<dim3(BB * TT / 16, CC / 64), 32, 0, stream>>>(Yb, WpT, bp, out);
}
